// MultiHeadedAttention_13108240187881
// MI455X (gfx1250) — compile-verified, matrix operand order checked by reading, not run on hardware
//
#include <hip/hip_runtime.h>

// ---------------------------------------------------------------------------
// MI455X / gfx1250 relation-aware multi-head attention.
// - relK/relV never materialized in d_model width (only 32-dim hiddens, f16)
// - softmax-invariant score terms dropped (s3, bias parts of s2/s4)
// - all dense GEMMs on v_wmma_f32_16x16x32_f16
// - B operands pre-packed to fragment-native layout -> b128 loads
// - edge tensor staged to LDS via the Tensor Data Mover (tensor_load_to_lds)
// ---------------------------------------------------------------------------

typedef __attribute__((ext_vector_type(16))) _Float16     v16h;
typedef __attribute__((ext_vector_type(8)))  _Float16     h8;
typedef __attribute__((ext_vector_type(8)))  float        v8f;
typedef __attribute__((ext_vector_type(4)))  unsigned int v4u;
typedef __attribute__((ext_vector_type(8)))  int          v8i_;
typedef __attribute__((ext_vector_type(4)))  int          v4i_;

#define BB 2
#define NN 512
#define HH 8
#define DKK 32
#define DMM 256
#define DEE 40
#define INV_SQRT_DK 0.17677669529663687f

__device__ __forceinline__ v8f wmma16(v16h a, v16h b, v8f c) {
  return __builtin_amdgcn_wmma_f32_16x16x32_f16(
      false, a, false, b, (short)0, c, false, false);
}

// ISA K-slot pattern for 16-bit A/B fragments:
// lane L: kb = 8*(L>>4); slot s -> K = kb + (s&8 ? 16 : 0) + (s&7)
__device__ __forceinline__ int kpat(int lane, int s) {
  return ((lane >> 4) << 3) + ((s & 8) ? 16 : 0) + (s & 7);
}

// A fragment (16x32 f16) from row-major source, ld in elements (16B aligned).
__device__ __forceinline__ v16h frag_rowmajor(const _Float16* base, int ld) {
  int l = threadIdx.x & 31;
  const _Float16* p = base + (size_t)(l & 15) * ld + ((l >> 4) << 3);
  h8 x0 = *(const h8*)p;
  h8 x1 = *(const h8*)(p + 16);
  v16h f;
#pragma unroll
  for (int s = 0; s < 8; ++s) { f[s] = x0[s]; f[8 + s] = x1[s]; }
  return f;
}

// B fragment from pre-packed buffer: [tile][lane][16 halves] contiguous.
__device__ __forceinline__ v16h frag_packed(const _Float16* pk) {
  int l = threadIdx.x & 31;
  const h8* p = (const h8*)(pk + (size_t)l * 16);
  h8 x0 = p[0], x1 = p[1];
  v16h f;
#pragma unroll
  for (int s = 0; s < 8; ++s) { f[s] = x0[s]; f[8 + s] = x1[s]; }
  return f;
}

// --------------------------- f32 -> f16 convert -----------------------------
__global__ void cvt_f16_kernel(const float* __restrict__ src,
                               _Float16* __restrict__ dst, int n) {
  int i = blockIdx.x * 256 + threadIdx.x;
  if (i < n) dst[i] = (_Float16)src[i];
}

// ------------------- pack B [K,N] into fragment-native layout ---------------
// out[((kt*(N/16)+nt)*32 + lane)*16 + s] = W[(kt*32 + kpat(lane,s))*N + nt*16 + (lane&15)]
__global__ void pack_bf32_kernel(const float* __restrict__ W,
                                 _Float16* __restrict__ out, int K, int N) {
  int idx = blockIdx.x * 256 + threadIdx.x;
  if (idx >= K * N) return;
  int s = idx & 15, lane = (idx >> 4) & 31, t = idx >> 9;
  int ntiles = N >> 4;
  int nt = t % ntiles, kt = t / ntiles;
  int k = kt * 32 + kpat(lane, s);
  int n = nt * 16 + (lane & 15);
  out[idx] = (_Float16)W[(size_t)k * N + n];
}

__global__ void pack_bf16_kernel(const _Float16* __restrict__ W,
                                 _Float16* __restrict__ out, int K, int N) {
  int idx = blockIdx.x * 256 + threadIdx.x;
  if (idx >= K * N) return;
  int s = idx & 15, lane = (idx >> 4) & 31, t = idx >> 9;
  int ntiles = N >> 4;
  int nt = t % ntiles, kt = t / ntiles;
  int k = kt * 32 + kpat(lane, s);
  int n = nt * 16 + (lane & 15);
  out[idx] = W[(size_t)k * N + n];
}

// pack W1 [40,32] zero-padded to K=64 -> 4 tiles (kt 0..1, nt 0..1), 2048 halves
__global__ void pack_w1_kernel(const float* __restrict__ W,
                               _Float16* __restrict__ out) {
  int idx = blockIdx.x * 256 + threadIdx.x;
  if (idx >= 2048) return;
  int s = idx & 15, lane = (idx >> 4) & 31, t = idx >> 9;
  int nt = t & 1, kt = t >> 1;
  int k = kt * 32 + kpat(lane, s);
  int n = nt * 16 + (lane & 15);
  out[idx] = (_Float16)((k < DEE) ? W[k * 32 + n] : 0.0f);
}

// ------------------ WMMA GEMM  C = A * Bpacked + bias -----------------------
// block = 256 (8 waves), wave -> one 16x16 tile, grid = (M/16, N/128)
__global__ void gemm_f16_kernel(const _Float16* __restrict__ A,
                                const _Float16* __restrict__ Bpk,
                                const float* __restrict__ bias,
                                _Float16* __restrict__ Cf16,
                                float* __restrict__ Cf32,
                                int M, int N, int K) {
  int wave = threadIdx.x >> 5;
  int m0 = blockIdx.x * 16;
  int nt = blockIdx.y * 8 + wave;
  int ntiles = N >> 4;
  v8f c = {};
  for (int k0 = 0; k0 < K; k0 += 32) {
    v16h a = frag_rowmajor(A + (size_t)m0 * K + k0, K);
    v16h b = frag_packed(Bpk + (size_t)((k0 >> 5) * ntiles + nt) * 512);
    c = wmma16(a, b, c);
  }
  int l = threadIdx.x & 31;
  int col = nt * 16 + (l & 15);
  float bb = bias ? bias[col] : 0.0f;
#pragma unroll
  for (int r = 0; r < 8; ++r) {
    int row = m0 + r + 8 * (l >> 4);
    float v = c[r] + bb;
    if (Cf16) Cf16[(size_t)row * N + col] = (_Float16)v;
    if (Cf32) Cf32[(size_t)row * N + col] = v;
  }
}

// ---------------- score-side coefficient vectors ----------------------------
// coef[b,h,i,c] = sum_d rkW2[c,h*32+d] * ( inv*(q+k)[b,h,i,d] + v[h,d] )
__global__ void coef_kernel(const _Float16* __restrict__ qf,
                            const _Float16* __restrict__ kf,
                            const float* __restrict__ rkW2,
                            const float* __restrict__ vvec,
                            float* __restrict__ coef) {
  int idx = blockIdx.x * 256 + threadIdx.x;  // B*H*N*32
  int c = idx & 31;
  int i = (idx >> 5) & (NN - 1);
  int h = (idx >> 14) & (HH - 1);
  int b = idx >> 17;
  const h8* qp = (const h8*)(qf + ((size_t)b * NN + i) * DMM + h * DKK);
  const h8* kp = (const h8*)(kf + ((size_t)b * NN + i) * DMM + h * DKK);
  const float* w = rkW2 + (size_t)c * DMM + h * DKK;
  const float* vv = vvec + h * DKK;
  float acc = 0.f;
#pragma unroll
  for (int g = 0; g < 4; ++g) {
    h8 qv = qp[g], kv = kp[g];
#pragma unroll
    for (int s = 0; s < 8; ++s) {
      int d = g * 8 + s;
      acc += w[d] * (INV_SQRT_DK * ((float)qv[s] + (float)kv[s]) + vv[d]);
    }
  }
  coef[idx] = acc;
}

// ---------------- edge MLP hiddens: TDM stage + WMMA ------------------------
// hidden(b,i,j,:) = leaky( pe[b,:,i,j] @ W1 + b1 )  for rk and rv.
// Block handles (b, i, 128 j). The [40 x 128] f32 edge tile (channel stride
// n^2) is DMA'd into LDS by the Tensor Data Mover, then 8 waves each run a
// 16-j WMMA tile. grid = (4, 512, 2), block = 256.
__global__ void hidden_kernel(const float* __restrict__ pe,
                              const float* __restrict__ rkb1,
                              const float* __restrict__ rvb1,
                              const _Float16* __restrict__ w1pk_k,
                              const _Float16* __restrict__ w1pk_v,
                              _Float16* __restrict__ hk,
                              _Float16* __restrict__ hv) {
  __shared__ float etile[DEE * 128];  // 20 KB TDM destination
  int tid = threadIdx.x;
  int wave = tid >> 5, l = tid & 31;
  int jb = blockIdx.x, i = blockIdx.y, b = blockIdx.z;
  int j0 = jb * 128;

  if (tid < 32) {  // wave 0 issues one TDM descriptor for the whole block
    unsigned lds_off =
        (unsigned)(unsigned long long)(__attribute__((address_space(3))) float*)&etile[0];
    unsigned long long ga = (unsigned long long)(uintptr_t)(
        pe + ((size_t)b * DEE * NN + i) * NN + j0);
    v4u g0;
    g0[0] = 1u;                                   // count=1 (valid user D#)
    g0[1] = lds_off;                              // lds_addr
    g0[2] = (unsigned)(ga & 0xFFFFFFFFu);         // global_addr[31:0]
    g0[3] = (unsigned)((ga >> 32) & 0x1FFFFFFu)   // global_addr[56:32]
            | (2u << 30);                         // type=2 (image)
    v8i_ g1;
    g1[0] = (int)(2u << 16);                      // data_size=4B code 2
    g1[1] = (int)(128u << 16);                    // tensor_dim0[15:0]=128
    g1[2] = (int)((unsigned)DEE << 16);           // tensor_dim1[15:0]=40
    g1[3] = (int)(128u << 16);                    // tile_dim0=128
    g1[4] = DEE;                                  // tile_dim1=40, tile_dim2=0
    g1[5] = NN * NN;                              // tensor_dim0_stride lo32
    g1[6] = 0;
    g1[7] = 0;
    v4i_ g2 = {0, 0, 0, 0}, g3 = {0, 0, 0, 0};    // 2D tensor: groups 2/3 unused
    v8i_ g4 = {0, 0, 0, 0, 0, 0, 0, 0};           // extra group (6-arg builtin)
    __builtin_amdgcn_tensor_load_to_lds(g0, g1, g2, g3, g4, 0);
    __builtin_amdgcn_s_wait_tensorcnt(0);
  }
  __syncthreads();

  // A fragments from LDS: rows = local j (16 per wave), K = channel (pad->64)
  int m = l & 15;
  int jl = wave * 16 + m;
  v16h a0, a1;
#pragma unroll
  for (int s = 0; s < 16; ++s) {
    int c0 = kpat(l, s);
    int c1 = c0 + 32;
    a0[s] = (_Float16)etile[c0 * 128 + jl];
    a1[s] = (_Float16)((c1 < DEE) ? etile[c1 * 128 + jl] : 0.0f);
  }

  int j0w = j0 + wave * 16;
#pragma unroll
  for (int mat = 0; mat < 2; ++mat) {
    const _Float16* pk = mat ? w1pk_v : w1pk_k;
    const float* bv1 = mat ? rvb1 : rkb1;
    _Float16* out = mat ? hv : hk;
#pragma unroll
    for (int nt = 0; nt < 2; ++nt) {
      v8f c = {};
      c = wmma16(a0, frag_packed(pk + (size_t)(0 * 2 + nt) * 512), c);
      c = wmma16(a1, frag_packed(pk + (size_t)(1 * 2 + nt) * 512), c);
      int cc = nt * 16 + (l & 15);
      float bias = bv1[cc];
#pragma unroll
      for (int r = 0; r < 8; ++r) {
        int j = j0w + r + 8 * (l >> 4);
        float v = c[r] + bias;
        v = v > 0.f ? v : 0.1f * v;  // LeakyReLU(0.1)
        out[(((size_t)b * NN + i) * NN + j) * DKK + cc] = (_Float16)v;
      }
    }
  }
}

// ---------------- scores + softmax -> p (f16) -------------------------------
// score = inv*(q_i . k_j) + coef[b,h,i,:] . hidden_k[b,i,j,:]
__global__ void score_softmax_kernel(const _Float16* __restrict__ qf,
                                     const _Float16* __restrict__ kf,
                                     const float* __restrict__ coef,
                                     const _Float16* __restrict__ hk,
                                     const int* __restrict__ mask,
                                     _Float16* __restrict__ p) {
  __shared__ float s[16][NN];
  __shared__ float cl[16][DKK];
  int it = blockIdx.x, h = blockIdx.y, b = blockIdx.z;
  int i0 = it * 16;
  int tid = threadIdx.x, wave = tid >> 5, l = tid & 31;

  for (int t = tid; t < 16 * DKK; t += 256) {
    int il = t >> 5, c = t & 31;
    cl[il][c] = coef[((size_t)((b * HH + h) * NN) + i0 + il) * DKK + c];
  }
  __syncthreads();

  v16h a = frag_rowmajor(qf + ((size_t)b * NN + i0) * DMM + h * DKK, DMM);

  for (int jt = wave; jt < NN / 16; jt += 8) {
    int j0 = jt * 16;
    v16h bf = frag_rowmajor(kf + ((size_t)b * NN + j0) * DMM + h * DKK, DMM);
    v8f c = {};
    c = wmma16(a, bf, c);
    int jl = l & 15;
    int j = j0 + jl;
    int mk = mask[b * NN + j];
    const _Float16* hpj = hk + (((size_t)b * NN + i0) * NN + j) * DKK;
#pragma unroll
    for (int r = 0; r < 8; ++r) {
      int il = r + 8 * (l >> 4);
      const h8* hh = (const h8*)(hpj + (size_t)il * NN * DKK);
      h8 x0 = hh[0], x1 = hh[1], x2 = hh[2], x3 = hh[3];
      float t2 = 0.f;
#pragma unroll
      for (int q2 = 0; q2 < 8; ++q2)
        t2 += cl[il][q2] * (float)x0[q2] + cl[il][8 + q2] * (float)x1[q2] +
              cl[il][16 + q2] * (float)x2[q2] + cl[il][24 + q2] * (float)x3[q2];
      float sc = INV_SQRT_DK * c[r] + t2;
      if (mk == 0) sc = -1e12f;
      s[il][j] = sc;
    }
  }
  __syncthreads();

  for (int il = wave * 2; il < wave * 2 + 2; ++il) {
    float mx = -3.4e38f;
    for (int j = l; j < NN; j += 32) mx = fmaxf(mx, s[il][j]);
#pragma unroll
    for (int off = 16; off > 0; off >>= 1) mx = fmaxf(mx, __shfl_xor(mx, off, 32));
    float sum = 0.f;
    for (int j = l; j < NN; j += 32) {
      float e = __expf(s[il][j] - mx);
      s[il][j] = e;
      sum += e;
    }
#pragma unroll
    for (int off = 16; off > 0; off >>= 1) sum += __shfl_xor(sum, off, 32);
    float rs = 1.0f / sum;
    _Float16* pp = p + ((size_t)(b * HH + h) * NN + i0 + il) * NN;
    for (int j = l; j < NN; j += 32) pp[j] = (_Float16)(s[il][j] * rs);
  }
}

// ---------------- o = p@val (WMMA) + rvW2^T (p@hidden_v) + rvb2 -------------
__global__ void out_attn_kernel(const _Float16* __restrict__ p,
                                const _Float16* __restrict__ vpk,  // packed V
                                const _Float16* __restrict__ hv,
                                const float* __restrict__ rvW2,
                                const float* __restrict__ rvb2,
                                _Float16* __restrict__ o) {
  __shared__ float cpart[8][16][16];
  __shared__ float ph[16][DKK];
  int it = blockIdx.x, h = blockIdx.y, b = blockIdx.z;
  int i0 = it * 16;
  int tid = threadIdx.x, wave = tid >> 5, l = tid & 31;
  int ntile = wave & 1, kg = wave >> 1;

  const _Float16* prow = p + ((size_t)(b * HH + h) * NN + i0) * NN;
  const _Float16* vb = vpk + (size_t)b * NN * DMM;
  v8f c = {};
#pragma unroll
  for (int ks = 0; ks < 4; ++ks) {
    int k0 = (kg * 4 + ks) * 32;  // j offset
    v16h a = frag_rowmajor(prow + k0, NN);
    v16h bf = frag_packed(vb + (size_t)((k0 >> 5) * 16 + (h * 2 + ntile)) * 512);
    c = wmma16(a, bf, c);
  }
  {
    int col = l & 15;
#pragma unroll
    for (int r = 0; r < 8; ++r) cpart[wave][r + 8 * (l >> 4)][col] = c[r];
  }

  // ph[i][c] = sum_j p[i,j] * hidden_v[i,j,c]
  int cc = tid & 31, ig = tid >> 5;
  const _Float16* pr0 = prow + (size_t)ig * NN;
  const _Float16* pr1 = pr0 + (size_t)8 * NN;
  const _Float16* hv0 = hv + (((size_t)b * NN + i0 + ig) * NN) * DKK + cc;
  const _Float16* hv1 = hv0 + (size_t)8 * NN * DKK;
  float acc0 = 0.f, acc1 = 0.f;
  for (int j = 0; j < NN; ++j) {
    acc0 += (float)pr0[j] * (float)hv0[(size_t)j * DKK];
    acc1 += (float)pr1[j] * (float)hv1[(size_t)j * DKK];
  }
  ph[ig][cc] = acc0;
  ph[ig + 8][cc] = acc1;
  __syncthreads();

  for (int t = tid; t < 16 * DKK; t += 256) {
    int il = t >> 5, d = t & 31;
    int dt = d >> 4, dc = d & 15;
    float v = rvb2[h * DKK + d];  // sum_j p == 1
    v += cpart[dt][il][dc] + cpart[dt + 2][il][dc] +
         cpart[dt + 4][il][dc] + cpart[dt + 6][il][dc];
    float s2 = 0.f;
#pragma unroll 8
    for (int c2 = 0; c2 < DKK; ++c2)
      s2 += rvW2[(size_t)c2 * DMM + h * DKK + d] * ph[il][c2];
    o[((size_t)b * NN + i0 + il) * DMM + h * DKK + d] = (_Float16)(v + s2);
  }
}

// ---------------------------------------------------------------------------
extern "C" void kernel_launch(void* const* d_in, const int* in_sizes, int n_in,
                              void* d_out, int out_size, void* d_ws, size_t ws_size,
                              hipStream_t stream) {
  const float* query = (const float*)d_in[0];
  const float* key_  = (const float*)d_in[1];
  const float* value = (const float*)d_in[2];
  const float* edges = (const float*)d_in[3];
  const int*   mask  = (const int*)d_in[4];
  const float* Wq = (const float*)d_in[5];  const float* bq = (const float*)d_in[6];
  const float* Wk = (const float*)d_in[7];  const float* bk = (const float*)d_in[8];
  const float* Wv = (const float*)d_in[9];  const float* bv = (const float*)d_in[10];
  const float* Wo = (const float*)d_in[11]; const float* bo = (const float*)d_in[12];
  const float* rkW1 = (const float*)d_in[13]; const float* rkb1 = (const float*)d_in[14];
  const float* rkW2 = (const float*)d_in[15]; // rkb2: softmax-invariant, dropped
  const float* rvW1 = (const float*)d_in[17]; const float* rvb1 = (const float*)d_in[18];
  const float* rvW2 = (const float*)d_in[19]; const float* rvb2 = (const float*)d_in[20];
  // u (d_in[21]): softmax-invariant, dropped
  const float* vvec = (const float*)d_in[22];
  float* out = (float*)d_out;

  const size_t XN = (size_t)BB * NN * DMM;  // 262144
  const size_t WN = (size_t)DMM * DMM;      // 65536
  char* ws = (char*)d_ws;
  _Float16* qin  = (_Float16*)ws; ws += XN * 2;
  _Float16* kin  = (_Float16*)ws; ws += XN * 2;
  _Float16* vin  = (_Float16*)ws; ws += XN * 2;
  _Float16* wqp  = (_Float16*)ws; ws += WN * 2;   // packed weights
  _Float16* wkp  = (_Float16*)ws; ws += WN * 2;
  _Float16* wvp  = (_Float16*)ws; ws += WN * 2;
  _Float16* wop  = (_Float16*)ws; ws += WN * 2;
  _Float16* w1pk = (_Float16*)ws; ws += 2048 * 2; // packed rkW1 (padded)
  _Float16* w1pv = (_Float16*)ws; ws += 2048 * 2; // packed rvW1 (padded)
  _Float16* qf   = (_Float16*)ws; ws += XN * 2;
  _Float16* kf   = (_Float16*)ws; ws += XN * 2;
  _Float16* valf = (_Float16*)ws; ws += XN * 2;
  _Float16* vpk  = (_Float16*)ws; ws += XN * 2;   // packed V for p@V
  _Float16* of   = (_Float16*)ws; ws += XN * 2;
  _Float16* hk   = (_Float16*)ws; ws += (size_t)BB * NN * NN * DKK * 2;
  _Float16* hv   = (_Float16*)ws; ws += (size_t)BB * NN * NN * DKK * 2;
  float*    coef = (float*)ws;    ws += (size_t)BB * HH * NN * DKK * 4;
  _Float16* pf   = (_Float16*)ws; ws += (size_t)BB * HH * NN * NN * 2;

  // 0) activations f32 -> f16; weights packed directly from f32
  cvt_f16_kernel<<<(XN + 255) / 256, 256, 0, stream>>>(query, qin, (int)XN);
  cvt_f16_kernel<<<(XN + 255) / 256, 256, 0, stream>>>(key_, kin, (int)XN);
  cvt_f16_kernel<<<(XN + 255) / 256, 256, 0, stream>>>(value, vin, (int)XN);
  pack_bf32_kernel<<<(WN + 255) / 256, 256, 0, stream>>>(Wq, wqp, DMM, DMM);
  pack_bf32_kernel<<<(WN + 255) / 256, 256, 0, stream>>>(Wk, wkp, DMM, DMM);
  pack_bf32_kernel<<<(WN + 255) / 256, 256, 0, stream>>>(Wv, wvp, DMM, DMM);
  pack_bf32_kernel<<<(WN + 255) / 256, 256, 0, stream>>>(Wo, wop, DMM, DMM);
  pack_w1_kernel<<<8, 256, 0, stream>>>(rkW1, w1pk);
  pack_w1_kernel<<<8, 256, 0, stream>>>(rvW1, w1pv);

  // 1) QKV projections (WMMA)
  dim3 gGemm(BB * NN / 16, DMM / 128);
  gemm_f16_kernel<<<gGemm, 256, 0, stream>>>(qin, wqp, bq, qf, nullptr, BB * NN, DMM, DMM);
  gemm_f16_kernel<<<gGemm, 256, 0, stream>>>(kin, wkp, bk, kf, nullptr, BB * NN, DMM, DMM);
  gemm_f16_kernel<<<gGemm, 256, 0, stream>>>(vin, wvp, bv, valf, nullptr, BB * NN, DMM, DMM);

  // 1b) repack V per batch into fragment layout for p@V
  for (int b = 0; b < BB; ++b)
    pack_bf16_kernel<<<(NN * DMM + 255) / 256, 256, 0, stream>>>(
        valf + (size_t)b * NN * DMM, vpk + (size_t)b * NN * DMM, NN, DMM);

  // 2) coef (folds s2+s4 weight contractions + inv scaling)
  coef_kernel<<<(BB * HH * NN * DKK) / 256, 256, 0, stream>>>(qf, kf, rkW2, vvec, coef);

  // 3) edge MLP hiddens: TDM stage + WMMA
  hidden_kernel<<<dim3(NN / 128, NN, BB), 256, 0, stream>>>(
      edges, rkb1, rvb1, w1pk, w1pv, hk, hv);

  // 4) scores + softmax
  score_softmax_kernel<<<dim3(NN / 16, HH, BB), 256, 0, stream>>>(qf, kf, coef, hk, mask, pf);

  // 5) attention output + relV correction
  out_attn_kernel<<<dim3(NN / 16, HH, BB), 256, 0, stream>>>(pf, vpk, hv, rvW2, rvb2, of);

  // 6) output projection -> f32
  gemm_f16_kernel<<<gGemm, 256, 0, stream>>>(of, wop, bo, nullptr, out, BB * NN, DMM, DMM);

  (void)in_sizes; (void)n_in; (void)out_size; (void)ws_size;
}
